// SpikformerCCSSA_QKFusion_ST_88983132438995
// MI455X (gfx1250) — matrix operand order repaired and checked by reading, not compiled, not run on hardware
//
#include <hip/hip_runtime.h>

// ---------------------------------------------------------------------------
// Spikformer spatial+temporal SSA, CDNA5 (gfx1250) wave32 WMMA implementation.
// LIF forward is the hard spike => all post-first-stage matmul inputs are
// binary {0,1}; everything runs in f16 with f32 WMMA accumulation.
// Attention is re-associated:  O = (V·Kᵀ)·Q  (spatial),  O = Q·(Kᵀ·V)
// (temporal), so the N×N / T×T attention matrices are never materialized.
// ---------------------------------------------------------------------------

typedef __attribute__((ext_vector_type(16))) _Float16 v16h;
typedef __attribute__((ext_vector_type(8)))  float    v8f;
typedef __attribute__((ext_vector_type(8)))  _Float16 h8;

#define TT  10
#define BBN 4
#define CC  384
#define NN  384
#define HH  12
#define DDm 32
#define TB  (TT * BBN)                 // 40
#define ELEMS ((size_t)TB * CC * NN)   // 5,898,240 per (T,B,C,N) tensor

// ---------------- WMMA fragment loaders (wave32, 16x16x32 f16) -------------
// A (16x32, M x K): lane L holds row m=L&15; halves 0..15 = K[kb..kb+15],
// kb = k0 + (L>=16 ? 16 : 0). Row-major A
// gives two contiguous 16-byte loads per lane.
__device__ __forceinline__ v16h wmma_load_a(const _Float16* __restrict__ A,
                                            int lda, int row0, int k0) {
  const int lane = threadIdx.x & 31;
  const int m  = row0 + (lane & 15);
  const int kb = k0 + ((lane >> 4) << 4);
  const _Float16* p = A + (size_t)m * lda + kb;
  h8 lo = *(const h8*)(p);
  h8 hi = *(const h8*)(p + 8);
  v16h r;
#pragma unroll
  for (int j = 0; j < 8; ++j) { r[j] = lo[j]; r[j + 8] = hi[j]; }
  return r;
}

// B (32x16, K x N) from a row-major K x N matrix: lane L holds column
// n = col0 + (L&15); halves j = B[kb + j][n], kb = k0 + (L>=16 ? 16 : 0).
// Column gather (2-byte strided loads).
__device__ __forceinline__ v16h wmma_load_b_gather(const _Float16* __restrict__ B,
                                                   int ldb, int k0, int col0) {
  const int lane = threadIdx.x & 31;
  const int n  = col0 + (lane & 15);
  const int kb = k0 + ((lane >> 4) << 4);
  v16h r;
#pragma unroll
  for (int j = 0; j < 16; ++j) r[j] = B[(size_t)(kb + j) * ldb + n];
  return r;
}

// B (32x16) where the logical B is the TRANSPOSE of a row-major matrix Kt:
// B[k][n] = Kt[n][k] -> per-lane contiguous along k (two 16-byte loads).
__device__ __forceinline__ v16h wmma_load_bT(const _Float16* __restrict__ Kt,
                                             int ldk, int k0, int col0) {
  const int lane = threadIdx.x & 31;
  const int n  = col0 + (lane & 15);
  const int kb = k0 + ((lane >> 4) << 4);
  const _Float16* p = Kt + (size_t)n * ldk + kb;
  h8 lo = *(const h8*)(p);
  h8 hi = *(const h8*)(p + 8);
  v16h r;
#pragma unroll
  for (int j = 0; j < 8; ++j) { r[j] = lo[j]; r[j + 8] = hi[j]; }
  return r;
}

__device__ __forceinline__ v8f wmma16(v16h a, v16h b, v8f c) {
  // D = A*B + C, f32 accumulate
  return __builtin_amdgcn_wmma_f32_16x16x32_f16(false, a, false, b,
                                                (short)0, c, false, false);
}

// --------------------------- elementwise helpers ---------------------------
__global__ void f32_to_f16_kernel(const float* __restrict__ in,
                                  _Float16* __restrict__ out, size_t n) {
  size_t i = (size_t)blockIdx.x * blockDim.x + threadIdx.x;
  if (i < n) out[i] = (_Float16)in[i];
}

// ---------------------- conv(1x1) + BN affine + LIF ------------------------
// Out[o,n] = spike( (sum_c W[o,c]*X[c,n]) * g[o] + b[o] )   spike: z >= 2.0
// One wave computes a 32x32 tile. Grid: (144 tiles, 40 tb).
__global__ __launch_bounds__(32)
void conv_lif_wmma(const _Float16* __restrict__ W,   // 384x384 row-major (o,c)
                   const _Float16* __restrict__ X,   // (TB, 384, 384)
                   const float* __restrict__ g,
                   const float* __restrict__ bsh,
                   _Float16* __restrict__ S) {       // (TB, 384, 384) spikes
  const int tile = blockIdx.x;           // 0..143
  const int tb   = blockIdx.y;           // 0..39
  const int tm = (tile / (NN / 32)) * 32;
  const int tn = (tile % (NN / 32)) * 32;
  const _Float16* Xb = X + (size_t)tb * CC * NN;
  _Float16*       Sb = S + (size_t)tb * CC * NN;

  v8f acc00 = {}, acc01 = {}, acc10 = {}, acc11 = {};
  for (int k0 = 0; k0 < CC; k0 += 32) {
    if (k0 + 32 < CC)  // hint: next B k-slab (lowers to global_prefetch_b8)
      __builtin_prefetch(Xb + (size_t)(k0 + 32) * NN + tn, 0, 1);
    v16h a0 = wmma_load_a(W, CC, tm,      k0);
    v16h a1 = wmma_load_a(W, CC, tm + 16, k0);
    v16h b0 = wmma_load_b_gather(Xb, NN, k0, tn);
    v16h b1 = wmma_load_b_gather(Xb, NN, k0, tn + 16);
    acc00 = wmma16(a0, b0, acc00);
    acc01 = wmma16(a0, b1, acc01);
    acc10 = wmma16(a1, b0, acc10);
    acc11 = wmma16(a1, b1, acc11);
  }

  const int lane = threadIdx.x & 31;
  const int colL = lane & 15;
  const int rh   = (lane >> 4) << 3;     // +0 / +8 row half
#pragma unroll
  for (int i = 0; i < 2; ++i) {
#pragma unroll
    for (int j = 0; j < 2; ++j) {
      const v8f& a = (i == 0) ? (j == 0 ? acc00 : acc01)
                              : (j == 0 ? acc10 : acc11);
#pragma unroll
      for (int r = 0; r < 8; ++r) {
        int row = tm + i * 16 + r + rh;
        int col = tn + j * 16 + colL;
        float z = a[r] * g[row] + bsh[row];
        Sb[(size_t)row * NN + col] = (z >= 2.0f) ? (_Float16)1.0f
                                                 : (_Float16)0.0f;
      }
    }
  }
}

// ----------------------- spatial attention + LIF ---------------------------
// Per (tb,h):  O = 0.25 * (V · Kᵀ) · Q, then spike (>= 8 pre-scale).
// Phase 1: Z(32x32) = V(32x384)·Kᵀ, all contiguous loads. Z -> LDS (f16).
// Phase 2: O(32x384) = Z·Q, B gathers from Q. One wave per block.
__global__ __launch_bounds__(32)
void spatial_attn_lif(const _Float16* __restrict__ Q,
                      const _Float16* __restrict__ K,
                      const _Float16* __restrict__ V,
                      _Float16* __restrict__ O) {
  __shared__ _Float16 Zs[32 * 48];       // ld=48 keeps h8 loads 16B aligned
  const int h  = blockIdx.x;             // 0..11
  const int tb = blockIdx.y;             // 0..39
  const size_t base = ((size_t)tb * CC + h * DDm) * NN;
  const _Float16* Qb = Q + base;
  const _Float16* Kb = K + base;
  const _Float16* Vb = V + base;
  _Float16*       Ob = O + base;

  // ---- Z = V · Kᵀ  (M=32, N=32, K=384) ----
  v8f z00 = {}, z01 = {}, z10 = {}, z11 = {};
  for (int k0 = 0; k0 < NN; k0 += 32) {
    v16h a0 = wmma_load_a(Vb, NN, 0,  k0);
    v16h a1 = wmma_load_a(Vb, NN, 16, k0);
    v16h b0 = wmma_load_bT(Kb, NN, k0, 0);
    v16h b1 = wmma_load_bT(Kb, NN, k0, 16);
    z00 = wmma16(a0, b0, z00);
    z01 = wmma16(a0, b1, z01);
    z10 = wmma16(a1, b0, z10);
    z11 = wmma16(a1, b1, z11);
  }
  const int lane = threadIdx.x & 31;
  const int colL = lane & 15;
  const int rh   = (lane >> 4) << 3;
#pragma unroll
  for (int i = 0; i < 2; ++i)
#pragma unroll
    for (int j = 0; j < 2; ++j) {
      const v8f& a = (i == 0) ? (j == 0 ? z00 : z01) : (j == 0 ? z10 : z11);
#pragma unroll
      for (int r = 0; r < 8; ++r)
        Zs[(size_t)(i * 16 + r + rh) * 48 + (j * 16 + colL)] = (_Float16)a[r];
    }
  __syncthreads();

  // ---- O = 0.25 * Z · Q  (M=32, N=384, K=32), LIF ----
  v16h a0 = wmma_load_a(&Zs[0], 48, 0,  0);
  v16h a1 = wmma_load_a(&Zs[0], 48, 16, 0);
  for (int n0 = 0; n0 < NN; n0 += 32) {
    v16h b0 = wmma_load_b_gather(Qb, NN, 0, n0);
    v16h b1 = wmma_load_b_gather(Qb, NN, 0, n0 + 16);
    v8f o00 = {}, o01 = {}, o10 = {}, o11 = {};
    o00 = wmma16(a0, b0, o00);
    o01 = wmma16(a0, b1, o01);
    o10 = wmma16(a1, b0, o10);
    o11 = wmma16(a1, b1, o11);
#pragma unroll
    for (int i = 0; i < 2; ++i)
#pragma unroll
      for (int j = 0; j < 2; ++j) {
        const v8f& a = (i == 0) ? (j == 0 ? o00 : o01) : (j == 0 ? o10 : o11);
#pragma unroll
        for (int r = 0; r < 8; ++r) {
          int row = i * 16 + r + rh;
          int col = n0 + j * 16 + colL;
          // spike( (0.25*acc)/TAU >= 1 )  <=>  acc >= 8
          Ob[(size_t)row * NN + col] = (a[r] >= 8.0f) ? (_Float16)1.0f
                                                      : (_Float16)0.0f;
        }
      }
  }
}

// ---------------------- temporal attention + LIF ---------------------------
// Per (n,b,h): O(TxD) = 0.25 * Q(TxD) · (Kᵀ·V)(DxD); T=10, D=32 (tiny).
// One wave per block, lane = output channel ee.
//
// NOTE on output placement: the reference does o.reshape(n,bb,T,DIM) on an
// (N,B,H,T,d) tensor, which reinterprets the flattened (H,T,d) index
// g = (h*T + t)*d + dd  as  (t2, c2) = (g / DIM, g % DIM) — i.e. heads are
// scrambled across time/channel. We reproduce that mapping exactly.
__global__ __launch_bounds__(32)
void temporal_attn_lif(const _Float16* __restrict__ Q,
                       const _Float16* __restrict__ K,
                       const _Float16* __restrict__ V,
                       _Float16* __restrict__ O) {
  const int n  = blockIdx.x;             // 0..383
  const int b  = blockIdx.y / HH;        // 0..3
  const int h  = blockIdx.y % HH;        // 0..11
  const int ee = threadIdx.x;            // 0..31

  float M[DDm];
#pragma unroll
  for (int dd = 0; dd < DDm; ++dd) M[dd] = 0.0f;

  // M[dd][ee] = sum_t K[t,dd] * V[t,ee]
  for (int t = 0; t < TT; ++t) {
    size_t rowb = (((size_t)t * BBN + b) * CC + h * DDm) * NN + n;
    float vt = (float)V[rowb + (size_t)ee * NN];
#pragma unroll
    for (int dd = 0; dd < DDm; ++dd)
      M[dd] += (float)K[rowb + (size_t)dd * NN] * vt;
  }
  // O[t,ee] = sum_dd Q[t,dd] * M[dd];  spike((0.25*s)/2 >= 1) <=> s >= 8
  for (int t = 0; t < TT; ++t) {
    size_t rowb = (((size_t)t * BBN + b) * CC + h * DDm) * NN + n;
    float s = 0.0f;
#pragma unroll
    for (int dd = 0; dd < DDm; ++dd)
      s += (float)Q[rowb + (size_t)dd * NN] * M[dd];
    _Float16 spike = (s >= 8.0f) ? (_Float16)1.0f : (_Float16)0.0f;
    // torch reshape scramble: g = (h*T + t)*d + ee -> (t2, c2)
    const int gidx = (h * TT + t) * DDm + ee;
    const int t2 = gidx / CC;
    const int c2 = gidx % CC;
    O[(((size_t)t2 * BBN + b) * CC + c2) * NN + n] = spike;
  }
}

// ------------------------------ reductions ---------------------------------
__global__ void reduce_mean_n_kernel(const _Float16* __restrict__ A,
                                     float* __restrict__ out) { // (40*384)
  int idx = blockIdx.x * blockDim.x + threadIdx.x;
  if (idx >= TB * CC) return;
  const _Float16* p = A + (size_t)idx * NN;
  float s = 0.0f;
  for (int i = 0; i < NN; ++i) s += (float)p[i];
  out[idx] = s * (1.0f / NN);
}

__global__ void reduce_mean_t_kernel(const _Float16* __restrict__ Bt,
                                     float* __restrict__ out) { // (4*384*384)
  size_t idx = (size_t)blockIdx.x * blockDim.x + threadIdx.x;
  const size_t per_t = (size_t)BBN * CC * NN;
  if (idx >= per_t) return;
  float s = 0.0f;
  for (int t = 0; t < TT; ++t) s += (float)Bt[(size_t)t * per_t + idx];
  out[idx] = s * (1.0f / TT);
}

__global__ void outer_final_kernel(const float* __restrict__ ar,  // (T,B,C)
                                   const float* __restrict__ br,  // (B,C,N)
                                   float* __restrict__ out) {     // (T,B,C,N)
  size_t i = (size_t)blockIdx.x * blockDim.x + threadIdx.x;
  if (i >= ELEMS) return;
  size_t n = i % NN;
  size_t c = (i / NN) % CC;
  size_t b = (i / ((size_t)NN * CC)) % BBN;
  size_t t = i / ((size_t)NN * CC * BBN);
  out[i] = ar[(t * BBN + b) * CC + c] * br[(b * CC + c) * NN + n];
}

// ------------------------------- launcher ----------------------------------
extern "C" void kernel_launch(void* const* d_in, const int* in_sizes, int n_in,
                              void* d_out, int out_size, void* d_ws,
                              size_t ws_size, hipStream_t stream) {
  (void)in_sizes; (void)n_in; (void)out_size; (void)ws_size;
  const float* x   = (const float*)d_in[0];
  const float* y   = (const float*)d_in[1];
  const float* sW  = (const float*)d_in[2];
  const float* sg  = (const float*)d_in[3];
  const float* sb  = (const float*)d_in[4];
  const float* tW  = (const float*)d_in[5];
  const float* tg  = (const float*)d_in[6];
  const float* tb  = (const float*)d_in[7];
  float* out = (float*)d_out;

  // Workspace layout (all offsets 256B aligned)
  char* ws = (char*)d_ws;
  size_t off = 0;
  _Float16* WH  = (_Float16*)(ws + off); off += (size_t)8 * CC * CC * 2;  // 8 f16 mats
  _Float16* XH  = (_Float16*)(ws + off); off += ELEMS * 2;
  _Float16* YH  = (_Float16*)(ws + off); off += ELEMS * 2;
  _Float16* QB  = (_Float16*)(ws + off); off += ELEMS * 2;
  _Float16* KB  = (_Float16*)(ws + off); off += ELEMS * 2;
  _Float16* VB  = (_Float16*)(ws + off); off += ELEMS * 2;
  _Float16* ATT = (_Float16*)(ws + off); off += ELEMS * 2;
  _Float16* CS  = (_Float16*)(ws + off); off += ELEMS * 2;
  float*    AR  = (float*)(ws + off);    off += (size_t)TB * CC * 4;
  float*    BR  = (float*)(ws + off);    off += (size_t)BBN * CC * NN * 4;

  const size_t WELEMS = (size_t)4 * CC * CC;     // one (4,384,384) weight pack
  const int TPB = 256;

  // ---- precision conversion ----
  f32_to_f16_kernel<<<(unsigned)((ELEMS + TPB - 1) / TPB), TPB, 0, stream>>>(x, XH, ELEMS);
  f32_to_f16_kernel<<<(unsigned)((ELEMS + TPB - 1) / TPB), TPB, 0, stream>>>(y, YH, ELEMS);
  f32_to_f16_kernel<<<(unsigned)((WELEMS + TPB - 1) / TPB), TPB, 0, stream>>>(sW, WH, WELEMS);
  f32_to_f16_kernel<<<(unsigned)((WELEMS + TPB - 1) / TPB), TPB, 0, stream>>>(tW, WH + WELEMS, WELEMS);

  const dim3 cgrid(144, TB);          // 32x32 tiles x (t,b)
  const size_t WM = (size_t)CC * CC;  // one 384x384 matrix

  // ---- spatial branch ----
  conv_lif_wmma<<<cgrid, 32, 0, stream>>>(WH + 0 * WM, XH, sg + 0,   sb + 0,   QB);
  conv_lif_wmma<<<cgrid, 32, 0, stream>>>(WH + 1 * WM, YH, sg + 384, sb + 384, KB);
  conv_lif_wmma<<<cgrid, 32, 0, stream>>>(WH + 2 * WM, YH, sg + 768, sb + 768, VB);
  spatial_attn_lif<<<dim3(HH, TB), 32, 0, stream>>>(QB, KB, VB, ATT);
  conv_lif_wmma<<<cgrid, 32, 0, stream>>>(WH + 3 * WM, ATT, sg + 1152, sb + 1152, CS);
  reduce_mean_n_kernel<<<(TB * CC + TPB - 1) / TPB, TPB, 0, stream>>>(CS, AR);

  // ---- temporal branch (reuse QB/KB/VB/ATT/CS) ----
  conv_lif_wmma<<<cgrid, 32, 0, stream>>>(WH + 4 * WM, XH, tg + 0,   tb + 0,   QB);
  conv_lif_wmma<<<cgrid, 32, 0, stream>>>(WH + 5 * WM, YH, tg + 384, tb + 384, KB);
  conv_lif_wmma<<<cgrid, 32, 0, stream>>>(WH + 6 * WM, YH, tg + 768, tb + 768, VB);
  temporal_attn_lif<<<dim3(NN, BBN * HH), 32, 0, stream>>>(QB, KB, VB, ATT);
  conv_lif_wmma<<<cgrid, 32, 0, stream>>>(WH + 7 * WM, ATT, tg + 1152, tb + 1152, CS);
  reduce_mean_t_kernel<<<(unsigned)(((size_t)BBN * CC * NN + TPB - 1) / TPB), TPB, 0, stream>>>(CS, BR);

  // ---- final outer product ----
  outer_final_kernel<<<(unsigned)((ELEMS + TPB - 1) / TPB), TPB, 0, stream>>>(AR, BR, out);
}
